// GPTQLinearAIU_60945585930578
// MI455X (gfx1250) — matrix operand order repaired and checked by reading, not compiled, not run on hardware
//
#include <hip/hip_runtime.h>

// GPTQ W4A16 linear for MI455X (gfx1250, wave32, WMMA + async global->LDS,
// 2-deep software pipeline: X tile double-buffered, qweight prefetched to regs).
// y[M,N] = x[M,K] * dequant(qweight)^T + bias
// M = B*S = 8192, K = IN = 4096, N = OUT = 4096, G = 32 (gsize = 128).

typedef __attribute__((ext_vector_type(16))) _Float16 v16h;
typedef __attribute__((ext_vector_type(8)))  _Float16 v8h;
typedef __attribute__((ext_vector_type(2)))  _Float16 v2h;
typedef __attribute__((ext_vector_type(8)))  float    v8f;
typedef __attribute__((ext_vector_type(4)))  int      v4i;

#define TILE_M 128
#define TILE_N 128
#define TILE_K 128               // == group size -> one (zero,scale) per row per tile
#define LDS_STRIDE (TILE_K + 8)  // +8 halves (16B) padding: conflict-free frag reads

union F16x16 { v16h v; v8h h[2]; };
union H2U    { unsigned int u; v2h h; };

// Async 16B global -> LDS copy (GLOBAL_LOAD_ASYNC_TO_LDS_B128, ASYNCcnt-tracked).
// LDS aperture flat addresses carry the wave-relative LDS byte offset in [31:0],
// which is exactly what the instruction's VDST (LDS address VGPR) expects.
__device__ __forceinline__ void async_copy_16B(const void* gptr, void* lptr) {
    asm volatile("global_load_async_to_lds_b128 %0, %1, off"
                 :: "v"((unsigned int)(unsigned long long)lptr),
                    "v"((unsigned long long)gptr)
                 : "memory");
}

// Wait until at most `n` async ops remain outstanding (per-wave ASYNCcnt).
__device__ __forceinline__ void wait_async_le8() {
    asm volatile("s_wait_asynccnt 0x8" ::: "memory");
}
__device__ __forceinline__ void wait_async_le0() {
    asm volatile("s_wait_asynccnt 0x0" ::: "memory");
}

__global__ __launch_bounds__(256, 1)
void gptq_w4a16_wmma_kernel(const _Float16* __restrict__ x,
                            const int*      __restrict__ qweight,
                            const int*      __restrict__ qzeros,
                            const _Float16* __restrict__ scales,
                            const _Float16* __restrict__ bias,
                            float*          __restrict__ y,
                            int M, int K, int N, int G)
{
    __shared__ __align__(16) _Float16 Xs[2][TILE_M * LDS_STRIDE];  // double-buffered
    __shared__ __align__(16) _Float16 Ws[TILE_N * LDS_STRIDE];     // single buffer

    const int tid  = threadIdx.x;
    const int lane = tid & 31;
    const int wave = tid >> 5;          // 0..7
    const int wm   = wave & 3;          // wave M index: rows wm*32 .. wm*32+31
    const int wn   = wave >> 2;         // wave N index: cols wn*64 .. wn*64+63

    const int block_n = blockIdx.x * TILE_N;
    const int block_m = blockIdx.y * TILE_M;

    // ---- X tile async-copy mapping: wave-coalesced 16B chunks ----
    const int xc  = (tid & 15) * 8;     // half offset within a row (16B chunk)
    const int xr0 = tid >> 4;           // starting row 0..15, step 16

    // ---- dequant mapping: thread pair per n-row, 8 int32 (64 nibbles) each ----
    const int dq_n   = tid >> 1;        // 0..127
    const int dq_kph = tid & 1;         // which half of the 16 packed ints
    const int dq_gn  = block_n + dq_n;  // global n-row this thread dequantizes
    const int* wq_base = qweight + (size_t)dq_gn * (K >> 3) + dq_kph * 8;

    v8f acc[2][4];
#pragma unroll
    for (int i = 0; i < 2; ++i)
#pragma unroll
        for (int j = 0; j < 4; ++j)
#pragma unroll
            for (int e = 0; e < 8; ++e)
                acc[i][j][e] = 0.0f;

    const int T = K / TILE_K;           // 32 K-tiles

    // ======== prologue: start tile 0 loads ========
#pragma unroll
    for (int rr = 0; rr < TILE_M; rr += 16) {
        const int row = xr0 + rr;
        async_copy_16B(x + (size_t)(block_m + row) * K + xc,
                       &Xs[0][row * LDS_STRIDE + xc]);
    }
    v4i rq0 = *(const v4i*)(wq_base);
    v4i rq1 = *(const v4i*)(wq_base + 4);
    unsigned int zr = (unsigned int)qzeros[(dq_gn >> 3)];          // g = 0
    _Float16     sc = scales[dq_gn];                               // g = 0

    for (int t = 0; t < T; ++t) {
        const int cur = t & 1;
        const bool has_next = (t + 1 < T);
        const int kt_next = (t + 1) * TILE_K;

        // ---- issue next tile's loads (in flight through this tile's WMMA) ----
        if (has_next) {
#pragma unroll
            for (int rr = 0; rr < TILE_M; rr += 16) {
                const int row = xr0 + rr;
                async_copy_16B(x + (size_t)(block_m + row) * K + kt_next + xc,
                               &Xs[cur ^ 1][row * LDS_STRIDE + xc]);
            }
        }
        v4i nq0, nq1; unsigned int nzr = 0; _Float16 nsc = (_Float16)0.0f;
        if (has_next) {
            const int* wqn = wq_base + (t + 1) * (TILE_K / 8);
            nq0 = *(const v4i*)(wqn);
            nq1 = *(const v4i*)(wqn + 4);
            nzr = (unsigned int)qzeros[(t + 1) * (N >> 3) + (dq_gn >> 3)];
            nsc = scales[(t + 1) * N + dq_gn];
        }

        // ---- dequantize W tile t (regs fetched one iteration ago) ----
        {
            const float zf = (float)((zr >> ((dq_gn & 7) * 4)) & 0xF);
            const float sf = (float)sc;
            // w = (nib - z)*s = (1024+nib)*s + (-(1024+z)*s); 0x6400|nib == fp16(1024+nib)
            const _Float16 sh = (_Float16)sf;
            const _Float16 ch = (_Float16)(-(1024.0f + zf) * sf);
            const v2h s2 = { sh, sh };
            const v2h c2 = { ch, ch };
#pragma unroll
            for (int j = 0; j < 8; ++j) {
                const unsigned int q =
                    (unsigned int)((j < 4) ? rq0[j & 3] : rq1[j & 3]);
                H2U p0, p1, p2, p3;
                p0.u = ( q         & 0x000F000Fu) | 0x64006400u;  // nib0, nib4
                p1.u = ((q >>  4)  & 0x000F000Fu) | 0x64006400u;  // nib1, nib5
                p2.u = ((q >>  8)  & 0x000F000Fu) | 0x64006400u;  // nib2, nib6
                p3.u = ((q >> 12)  & 0x000F000Fu) | 0x64006400u;  // nib3, nib7
                const v2h w0 = p0.h * s2 + c2;   // v_pk_fma_f16
                const v2h w1 = p1.h * s2 + c2;
                const v2h w2 = p2.h * s2 + c2;
                const v2h w3 = p3.h * s2 + c2;
                v8h out;                         // natural k order 0..7
                out[0] = w0[0]; out[1] = w1[0]; out[2] = w2[0]; out[3] = w3[0];
                out[4] = w0[1]; out[5] = w1[1]; out[6] = w2[1]; out[7] = w3[1];
                *(v8h*)(&Ws[dq_n * LDS_STRIDE + (dq_kph * 8 + j) * 8]) = out;
            }
        }
        // rotate prefetched regs for next iteration
        if (has_next) { rq0 = nq0; rq1 = nq1; zr = nzr; sc = nsc; }

        // X tile t resident (only tile t+1's 8 async ops may remain in flight)
        if (has_next) wait_async_le8(); else wait_async_le0();
        __syncthreads();

        // ======== WMMA: 4 k-steps of 32 over the 128 K-tile ========
        const _Float16* Xbuf = &Xs[cur][0];
#pragma unroll
        for (int ks = 0; ks < TILE_K; ks += 32) {
            // A fragments (16x32 f16): lane<16 row M=lane holds K {0..7,16..23};
            // lane>=16 row M=lane-16 holds K {8..15,24..31}.
            const int a_koff = ks + ((lane & 16) ? 8 : 0);
            F16x16 afrag[2];
#pragma unroll
            for (int i = 0; i < 2; ++i) {
                const _Float16* ap =
                    &Xbuf[(wm * 32 + i * 16 + (lane & 15)) * LDS_STRIDE + a_koff];
                afrag[i].h[0] = *(const v8h*)(ap);        // K +0..7
                afrag[i].h[1] = *(const v8h*)(ap + 16);   // K +16..23
            }
            // B fragments (32x16 f16): lane holds column N=lane%16,
            // K {0..7,16..23} (lanes 0-15) / {8..15,24..31} (lanes 16-31).
            const int b_koff = ks + ((lane & 16) ? 8 : 0);
            F16x16 bfrag[4];
#pragma unroll
            for (int j = 0; j < 4; ++j) {
                const _Float16* bp =
                    &Ws[(wn * 64 + j * 16 + (lane & 15)) * LDS_STRIDE + b_koff];
                bfrag[j].h[0] = *(const v8h*)(bp);        // K +0..7
                bfrag[j].h[1] = *(const v8h*)(bp + 16);   // K +16..23
            }
#pragma unroll
            for (int i = 0; i < 2; ++i)
#pragma unroll
                for (int j = 0; j < 4; ++j)
                    acc[i][j] = __builtin_amdgcn_wmma_f32_16x16x32_f16(
                        /*neg_a=*/false, afrag[i].v,
                        /*neg_b=*/false, bfrag[j].v,
                        /*c_mod=*/(short)0, acc[i][j],
                        /*reuse_a=*/false, /*reuse_b=*/false);
        }
        __syncthreads();   // retire all readers of Xs[cur] and Ws before overwrite
    }

    // ======== epilogue: C/D layout -> global, + bias ========
    // VGPR v: lanes 0-15 -> M = base + v, N = lane; lanes 16-31 -> M = base+8+v.
    const int col   = lane & 15;
    const int mhalf = (lane & 16) ? 8 : 0;
#pragma unroll
    for (int i = 0; i < 2; ++i) {
#pragma unroll
        for (int j = 0; j < 4; ++j) {
            const int n  = block_n + wn * 64 + j * 16 + col;
            const float bv = (float)bias[n];
            const int m0 = block_m + wm * 32 + i * 16 + mhalf;
            float* yp = y + (size_t)m0 * N + n;
#pragma unroll
            for (int v = 0; v < 8; ++v)
                yp[(size_t)v * N] = acc[i][j][v] + bv;
        }
    }
}

extern "C" void kernel_launch(void* const* d_in, const int* in_sizes, int n_in,
                              void* d_out, int out_size, void* d_ws, size_t ws_size,
                              hipStream_t stream) {
    (void)n_in; (void)out_size; (void)d_ws; (void)ws_size;
    const _Float16* x       = (const _Float16*)d_in[0];
    const int*      qweight = (const int*)d_in[1];
    const int*      qzeros  = (const int*)d_in[2];
    const _Float16* scales  = (const _Float16*)d_in[3];
    const _Float16* bias    = (const _Float16*)d_in[4];
    float*          y       = (float*)d_out;

    const int K = 4096, N = 4096, G = 32;
    const int M = in_sizes[0] / K;      // B*S = 8192

    dim3 grid(N / TILE_N, M / TILE_M);  // 32 x 64 blocks
    dim3 block(256);                    // 8 wave32 waves
    hipLaunchKernelGGL(gptq_w4a16_wmma_kernel, grid, block, 0, stream,
                       x, qweight, qzeros, scales, bias, y, M, K, N, G);
}
